// SelfAttentionLayer_69750268887504
// MI455X (gfx1250) — compile-verified, matrix operand order checked by reading, not run on hardware
//
#include <hip/hip_runtime.h>
#include <hip/hip_bf16.h>

typedef __attribute__((ext_vector_type(16))) _Float16 v16h;
typedef __attribute__((ext_vector_type(8)))  _Float16 v8h;
typedef __attribute__((ext_vector_type(4)))  _Float16 v4h;
typedef __attribute__((ext_vector_type(8)))  float    v8f;

#define STAGE_OFF  262144   // byte offset of weight staging buffers in LDS
#define STAGE_STRB 20480    // one staging buffer: 256 rows * 80 B
#define ROWPADH    40       // staged row stride in f16 units (80 B)

union Frag16 { v16h v; v8h h[2]; };

// CDNA5 f16 A(16x32)/B(32x16) fragment: lane L holds row/col L%16; K-chunks at
// kbase + 8*(L/16) .. +7 (VGPR0-3) and kbase + 16 + 8*(L/16) .. +7 (VGPR4-7).
__device__ __forceinline__ v16h load_frag(const _Float16* __restrict__ base,
                                          int row, int kbase, int sub8) {
  Frag16 f;
  const _Float16* p = base + row * 256 + kbase + sub8;
  f.h[0] = *(const v8h*)(p);
  f.h[1] = *(const v8h*)(p + 16);
  return f.v;
}

// B-fragment from a staged [256 n][32 c] chunk with 80-byte padded rows.
__device__ __forceinline__ v16h load_frag_w(const _Float16* __restrict__ base,
                                            int n, int sub8) {
  Frag16 f;
  const _Float16* p = base + n * ROWPADH + sub8;
  f.h[0] = *(const v8h*)(p);
  f.h[1] = *(const v8h*)(p + 16);
  return f.v;
}

__device__ __forceinline__ v8f wmma_f16(v16h a, v16h b, v8f c) {
  return __builtin_amdgcn_wmma_f32_16x16x32_f16(false, a, false, b,
                                                (short)0, c, false, false);
}

__device__ __forceinline__ float hmax16(float v) {
  v = fmaxf(v, __shfl_xor(v, 1, 16));
  v = fmaxf(v, __shfl_xor(v, 2, 16));
  v = fmaxf(v, __shfl_xor(v, 4, 16));
  v = fmaxf(v, __shfl_xor(v, 8, 16));
  return v;
}
__device__ __forceinline__ float hsum16(float v) {
  v += __shfl_xor(v, 1, 16);
  v += __shfl_xor(v, 2, 16);
  v += __shfl_xor(v, 4, 16);
  v += __shfl_xor(v, 8, 16);
  return v;
}

// Async-stage one [256 n][32 c] f16 weight chunk (16 KB) into LDS.
// Each of the 512 threads moves 2 x 16 B with global_load_async_to_lds_b128.
__device__ __forceinline__ void stage_chunk(const _Float16* __restrict__ wtm,
                                            int kkbytes, uint32_t ldsA,
                                            uint32_t ldsB, uint32_t voffA,
                                            uint32_t voffB) {
  uint64_t sb = (uint64_t)(uintptr_t)wtm;
  asm volatile(
      "global_load_async_to_lds_b128 %0, %2, %4\n\t"
      "global_load_async_to_lds_b128 %1, %3, %4"
      :
      : "v"(ldsA), "v"(ldsB), "v"(voffA + kkbytes), "v"(voffB + kkbytes),
        "s"(sb)
      : "memory");
}

__device__ __forceinline__ void wait_async_barrier() {
  asm volatile("s_wait_asynccnt 0x0" ::: "memory");
  __syncthreads();
}

// Staged GEMM: acc[8] += Xs[16 rows] * W_chunk over k = 0..255 with
// double-buffered async weight streaming; tbase selects the 8 n-tiles.
__device__ __forceinline__ void gemm_staged8(
    const _Float16* __restrict__ wtm, const _Float16* __restrict__ Xs,
    const char* smembase, int arow, int ln16, int sub8, int tbase,
    uint32_t lds0, uint32_t lds1, uint32_t voff0, uint32_t voff1,
    v8f acc[8]) {
  stage_chunk(wtm, 0, lds0, lds1, voff0, voff1);  // prime buf0 with chunk 0
  wait_async_barrier();
  #pragma unroll
  for (int kk = 0; kk < 8; ++kk) {
    const int buf = kk & 1;
    if (kk < 7)
      stage_chunk(wtm, (kk + 1) * 64, lds0 + (buf ^ 1) * STAGE_STRB,
                  lds1 + (buf ^ 1) * STAGE_STRB, voff0, voff1);
    const _Float16* ch =
        (const _Float16*)(smembase + STAGE_OFF + buf * STAGE_STRB);
    v16h a = load_frag(Xs, arow, kk * 32, sub8);
    v16h b = load_frag_w(ch, tbase * 16 + ln16, sub8);
    #pragma unroll
    for (int t = 0; t < 8; ++t) {       // rotate: load b(t+1) before wmma(t)
      v16h bn = (t < 7) ? load_frag_w(ch, (tbase + t + 1) * 16 + ln16, sub8) : b;
      acc[t] = wmma_f16(a, b, acc[t]);
      b = bn;
    }
    if (kk < 7) wait_async_barrier();
  }
}

// Convert + transpose weights to f16: wt[n][c] = w[c][n], for Wq, Wk, Wv.
__global__ void prep_w(const float* __restrict__ wq, const float* __restrict__ wk,
                       const float* __restrict__ wv, _Float16* __restrict__ wt) {
  int i = blockIdx.x * 256 + threadIdx.x;  // 0..65535, i = n*256 + c
  int n = i >> 8, c = i & 255;
  int src = c * 256 + n;
  wt[i]          = (_Float16)wq[src];
  wt[i + 65536]  = (_Float16)wk[src];
  wt[i + 131072] = (_Float16)wv[src];
}

__global__ __launch_bounds__(512) void attn_main(
    const float* __restrict__ x,
    const float* __restrict__ bq, const float* __restrict__ bk,
    const float* __restrict__ bv, const float* __restrict__ gamma,
    const _Float16* __restrict__ wt,   // [3][256 n][256 c] f16 (pre-transposed)
    float* __restrict__ out) {
  extern __shared__ char smem[];
  _Float16* Xs = (_Float16*)smem;              // X[w][c] f16; later V^T[c][j]
  _Float16* Ks = (_Float16*)(smem + 131072);   // Q-scratch -> K[j][c] -> P-scratch

  const int tid  = threadIdx.x;
  const int lane = tid & 31;
  const int wid  = tid >> 5;          // 16 waves
  const int ln16 = lane & 15;
  const int sub8 = (lane >> 4) * 8;   // 8*(lane/16)
  const int row0 = wid * 16;          // this wave's 16 pixel rows
  const int arow = row0 + ln16;

  // Per-thread async-staging addresses: thread moves chunk rows n=tid/4 and
  // n=tid/4+128, 16-byte quadrant (tid&3). Global rows are 512 B apart,
  // staged LDS rows are padded to 80 B (bank-conflict-free fragment reads).
  const uint32_t voff0 = (uint32_t)((tid >> 2) * 512 + (tid & 3) * 16);
  const uint32_t voff1 = voff0 + 128 * 512;
  const uint32_t lds0  = (uint32_t)(STAGE_OFF + (tid >> 2) * 80 + (tid & 3) * 16);
  const uint32_t lds1  = lds0 + 128 * 80;

  const long bh   = blockIdx.x;       // (b*H + h)
  const float* xr = x   + bh * 65536;
  float* outr     = out + bh * 65536;

  // ---- Stage X (f32 -> f16) into LDS, coalesced b128 loads ----
  for (int i = tid; i < 16384; i += 512) {
    float4 f = ((const float4*)xr)[i];
    v4h h = { (_Float16)f.x, (_Float16)f.y, (_Float16)f.z, (_Float16)f.w };
    *(v4h*)(Xs + i * 4) = h;
  }
  __syncthreads();

  const _Float16* wtq = wt;
  const _Float16* wtk = wt + 65536;
  const _Float16* wtv = wt + 131072;
  _Float16* slice = Ks + wid * 4096;  // wave's 16x256 region == its K rows

  // ================= Q = X*Wq + bq, relayout D->A fragments =================
  v16h qa[8];
  {
    #pragma unroll
    for (int half = 0; half < 2; ++half) {
      v8f acc[8] = {};
      gemm_staged8(wtq, Xs, smem, arow, ln16, sub8, half * 8,
                   lds0, lds1, voff0, voff1, acc);
      #pragma unroll
      for (int t = 0; t < 8; ++t) {
        int c = (half * 8 + t) * 16 + ln16;
        float bb = bq[c];
        #pragma unroll
        for (int d = 0; d < 8; ++d)
          slice[(sub8 + d) * 256 + c] = (_Float16)(acc[t][d] + bb);
      }
    }
    #pragma unroll
    for (int kk = 0; kk < 8; ++kk)     // read back as A fragments (in-order DS)
      qa[kk] = load_frag(slice, ln16, kk * 32, sub8);
  }

  // ================= K = X*Wk + bk -> Ks[j][c] (same per-wave slice) ========
  #pragma unroll
  for (int half = 0; half < 2; ++half) {
    v8f acc[8] = {};
    gemm_staged8(wtk, Xs, smem, arow, ln16, sub8, half * 8,
                 lds0, lds1, voff0, voff1, acc);
    #pragma unroll
    for (int t = 0; t < 8; ++t) {
      int c = (half * 8 + t) * 16 + ln16;
      float bb = bk[c];
      #pragma unroll
      for (int d = 0; d < 8; ++d)
        slice[(sub8 + d) * 256 + c] = (_Float16)(acc[t][d] + bb);
    }
  }

  // ================= V = X*Wv + bv, keep f16 in registers ===================
  v8h vh[16];
  #pragma unroll
  for (int half = 0; half < 2; ++half) {
    v8f acc[8] = {};
    gemm_staged8(wtv, Xs, smem, arow, ln16, sub8, half * 8,
                 lds0, lds1, voff0, voff1, acc);
    #pragma unroll
    for (int t = 0; t < 8; ++t) {
      int c = (half * 8 + t) * 16 + ln16;
      float bb = bv[c];
      v8h hh;
      #pragma unroll
      for (int d = 0; d < 8; ++d) hh[d] = (_Float16)(acc[t][d] + bb);
      vh[half * 8 + t] = hh;
    }
  }
  __syncthreads();   // all waves done reading Xs and writing K

  // V^T into Xs region: Vt[c][j]; 8 j-contiguous f16 per lane -> b128 store
  #pragma unroll
  for (int t = 0; t < 16; ++t) {
    int c = t * 16 + ln16;
    *(v8h*)(Xs + c * 256 + row0 + sub8) = vh[t];
  }
  __syncthreads();   // Vt + K fully visible

  // ================= S = Q * K^T (full 16 j-tiles live) =====================
  v8f s[16] = {};
  #pragma unroll
  for (int kk = 0; kk < 8; ++kk) {
    v16h b = load_frag(Ks, ln16, kk * 32, sub8);
    #pragma unroll
    for (int t = 0; t < 16; ++t) {      // rotate: load b(t+1) before wmma(t)
      v16h bn = (t < 15) ? load_frag(Ks, (t + 1) * 16 + ln16, kk * 32, sub8) : b;
      s[t] = wmma_f16(qa[kk], b, s[t]);
      b = bn;
    }
  }

  // ================= softmax over j (256) per row ===========================
  float inv[8];
  #pragma unroll
  for (int d = 0; d < 8; ++d) {
    float m = -3.4e38f;
    #pragma unroll
    for (int t = 0; t < 16; ++t) m = fmaxf(m, s[t][d]);
    m = hmax16(m);
    float sum = 0.f;
    #pragma unroll
    for (int t = 0; t < 16; ++t) { float e = __expf(s[t][d] - m); s[t][d] = e; sum += e; }
    sum = hsum16(sum);
    inv[d] = 1.0f / sum;
  }

  __syncthreads();   // all waves done reading Ks; reuse as P scratch
  #pragma unroll
  for (int t = 0; t < 16; ++t) {
    int j = t * 16 + ln16;
    #pragma unroll
    for (int d = 0; d < 8; ++d)
      slice[(sub8 + d) * 256 + j] = (_Float16)(s[t][d] * inv[d]);
  }
  // no barrier: each wave reads only its own slice (DS ops in-order per wave)

  // ================= O = P * V ; out = gamma*O + x ==========================
  #pragma unroll
  for (int half = 0; half < 2; ++half) {
    v8f o[8] = {};
    #pragma unroll
    for (int kk = 0; kk < 8; ++kk) {
      v16h a = load_frag(slice, ln16, kk * 32, sub8);   // P A-fragment
      v16h b = load_frag(Xs, (half * 8) * 16 + ln16, kk * 32, sub8);
      #pragma unroll
      for (int t = 0; t < 8; ++t) {     // rotate: load b(t+1) before wmma(t)
        v16h bn = (t < 7)
            ? load_frag(Xs, (half * 8 + t + 1) * 16 + ln16, kk * 32, sub8) : b;
        o[t] = wmma_f16(a, b, o[t]);
        b = bn;
      }
    }
    #pragma unroll
    for (int t = 0; t < 8; ++t) {
      int c = (half * 8 + t) * 16 + ln16;
      float g = gamma[c];
      #pragma unroll
      for (int d = 0; d < 8; ++d) {
        int r = row0 + sub8 + d;
        outr[r * 256 + c] = g * o[t][d] + xr[r * 256 + c];
      }
    }
  }
}

extern "C" void kernel_launch(void* const* d_in, const int* in_sizes, int n_in,
                              void* d_out, int out_size, void* d_ws, size_t ws_size,
                              hipStream_t stream) {
  const float* x     = (const float*)d_in[0];
  const float* wq    = (const float*)d_in[1];
  const float* bq    = (const float*)d_in[2];
  const float* wk    = (const float*)d_in[3];
  const float* bk    = (const float*)d_in[4];
  const float* wv    = (const float*)d_in[5];
  const float* bv    = (const float*)d_in[6];
  const float* gamma = (const float*)d_in[7];
  _Float16* wt = (_Float16*)d_ws;   // 3 * 256*256 * 2 B = 384 KB

  (void)hipFuncSetAttribute(reinterpret_cast<const void*>(attn_main),
                            hipFuncAttributeMaxDynamicSharedMemorySize, 303104);

  prep_w<<<256, 256, 0, stream>>>(wq, wk, wv, wt);
  attn_main<<<2048, 512, 303104, stream>>>(x, bq, bk, bv, gamma, wt,
                                           (float*)d_out);
}
